// GraphX_91147795956296
// MI455X (gfx1250) — matrix operand order repaired and checked by reading, not compiled, not run on hardware
//
#include <hip/hip_runtime.h>
#include <hip/hip_bf16.h>
#include <math.h>

#define F_N    20000
#define S_N    60000
#define U_N    2000
#define I_N    2000
#define NSUP_N (S_N + U_N + I_N)     // 64000
#define DEG    8
#define E_N    (NSUP_N * DEG)        // 512000
#define HID    128
#define NH     8
#define DH     16
#define SE_N   256
#define FFN_N  512

#define CDIV(a,b) (((a)+(b)-1)/(b))

typedef __bf16 v16bf __attribute__((ext_vector_type(16)));
typedef __bf16 v8bf  __attribute__((ext_vector_type(8)));
typedef float  v8f   __attribute__((ext_vector_type(8)));
typedef float  v4f   __attribute__((ext_vector_type(4)));
typedef unsigned int u32x4 __attribute__((ext_vector_type(4)));
typedef int    v8i   __attribute__((ext_vector_type(8)));
typedef int    v4i   __attribute__((ext_vector_type(4)));

// ---------------- helpers ----------------

__device__ __forceinline__ __bf16 f2bf(float x) {
    unsigned u = __float_as_uint(x);
    u += 0x7FFFu + ((u >> 16) & 1u);          // round-to-nearest-even
    unsigned short h = (unsigned short)(u >> 16);
    __bf16 r;
    __builtin_memcpy(&r, &h, 2);
    return r;
}

__device__ __forceinline__ void atomicMaxF(float* addr, float val) {
    if (val >= 0.f) atomicMax((int*)addr, __float_as_int(val));
    else            atomicMin((unsigned int*)addr, __float_as_uint(val));
}

// ---------------- weight packing (fp32 [K,N] -> bf16 WMMA B-fragment order) ----
// Fragment layout per CDNA5 ISA: lane holds col = n0+(lane&15),
// K rows kb..kb+7 then kb+16..kb+23 with kb = k0 + (lane>>4)*8.
// Packed: P[((kt*tilesN + nt)*32 + lane)*16 + e]
__global__ void k_pack_b(const float* __restrict__ B, __bf16* __restrict__ P,
                         int K, int N) {
    int t = blockIdx.x * blockDim.x + threadIdx.x;
    int tilesN = N >> 4;
    int total = (K >> 5) * tilesN * 32;
    if (t >= total) return;
    int lane = t & 31;
    int tile = t >> 5;
    int nt = tile % tilesN, kt = tile / tilesN;
    int col = nt * 16 + (lane & 15);
    int kb  = kt * 32 + ((lane >> 4) << 3);
    __bf16* o = P + (size_t)t * 16;
    #pragma unroll
    for (int e = 0; e < 16; ++e) o[e] = f2bf(B[(size_t)(kb + e + ((e >> 3) << 3)) * N + col]);
}

// pack per-head W[NH,HID,DH] as Wcat[d, h*16+e'] in B-fragment order (K=HID,N=HID)
__global__ void k_pack_Whead(const float* __restrict__ W, __bf16* __restrict__ P) {
    int t = blockIdx.x * blockDim.x + threadIdx.x;   // (HID/32)*(HID/16)*32 = 1024
    if (t >= (HID / 32) * (HID / 16) * 32) return;
    int lane = t & 31;
    int tile = t >> 5;
    int nt = tile & 7, kt = tile >> 3;               // nt == head
    int ep = lane & 15;
    int kb = kt * 32 + ((lane >> 4) << 3);
    __bf16* o = P + (size_t)t * 16;
    #pragma unroll
    for (int e = 0; e < 16; ++e)
        o[e] = f2bf(W[(size_t)nt * (HID * DH) + (size_t)(kb + e + ((e >> 3) << 3)) * DH + ep]);
}

// ---------------- generic bf16-WMMA GEMM (packed B) ----------------
// C[M,N] = act( (gather? A[idxA[m]] : A[m]) @ B + bias ); one wave per 16x16 tile.
__global__ void k_gemm_bf16(const float* __restrict__ A, const int* __restrict__ idxA,
                            int lda, const v16bf* __restrict__ Bp, int N,
                            const float* __restrict__ bias,
                            float* __restrict__ C, int M, int K, int act)
{
    const int lane = threadIdx.x & 31;
    const int wid  = threadIdx.x >> 5;
    const int tilesN = N >> 4;
    const int tiles  = (M >> 4) * tilesN;
    int tile = blockIdx.x * (blockDim.x >> 5) + wid;
    if (tile >= tiles) return;                        // wave-uniform exit
    const int tm = tile / tilesN, tn = tile - tm * tilesN;
    const int m0 = tm << 4, n0 = tn << 4;

    int arow = m0 + (lane & 15);
    if (idxA) arow = idxA[arow];
    const float* aptr = A + (size_t)arow * lda + ((lane >> 4) << 3);

    v8f acc = {};
    for (int k0 = 0; k0 < K; k0 += 32) {
        const v4f* ap4 = (const v4f*)(aptr + k0);
        v4f a0 = ap4[0], a1 = ap4[1], a2 = ap4[4], a3 = ap4[5];
        v16bf af;
        #pragma unroll
        for (int e = 0; e < 4; ++e) {
            af[e]      = f2bf(a0[e]);
            af[e + 4]  = f2bf(a1[e]);
            af[e + 8]  = f2bf(a2[e]);
            af[e + 12] = f2bf(a3[e]);
        }
        v16bf bfr = Bp[(size_t)((k0 >> 5) * tilesN + tn) * 32 + lane];
        acc = __builtin_amdgcn_wmma_f32_16x16x32_bf16(false, af, false, bfr,
                                                      (short)0, acc, false, false);
    }
    const int r0 = m0 + ((lane >> 4) << 3);
    const int c  = n0 + (lane & 15);
    const float bs = bias ? bias[c] : 0.f;
    #pragma unroll
    for (int r = 0; r < 8; ++r) {
        float v = acc[r] + bs;
        if (act == 1) v = fmaxf(v, 0.f);
        C[(size_t)(r0 + r) * N + c] = v;
    }
}

// ---------------- fused FFN: out = h + relu(ln@W1+b1)@W2 + b2 ----------------
// One block (8 waves) per 16-row tile. ln tile staged via Tensor Data Mover;
// 16x512 intermediate lives in LDS as bf16.
__global__ void k_ffn(const float* __restrict__ ln, const float* __restrict__ hres,
                      const v16bf* __restrict__ W1p, const float* __restrict__ b1,
                      const v16bf* __restrict__ W2p, const float* __restrict__ b2,
                      float* __restrict__ out)
{
    __shared__ __align__(16) float  lnF[16 * HID];
    __shared__ __align__(16) __bf16 mid[16 * FFN_N];
    const int lane = threadIdx.x & 31;
    const int wid  = threadIdx.x >> 5;     // 0..7
    const int m0   = blockIdx.x << 4;

#if __has_builtin(__builtin_amdgcn_tensor_load_to_lds) && __has_builtin(__builtin_amdgcn_s_wait_tensorcnt)
    if (wid == 0) {
        // D# for a contiguous 16x128 fp32 tile (8KB): 1 row of 2048 elements.
        unsigned long long ga = (unsigned long long)(const void*)(ln + (size_t)m0 * HID);
        unsigned ldsOff = (unsigned)(unsigned long long)(void*)&lnF[0];
        u32x4 g0; v8i g1; v4i gz = {};
        g0[0] = 1u;                                           // count=1 (valid D#)
        g0[1] = ldsOff;                                       // lds_addr
        g0[2] = (unsigned)(ga & 0xFFFFFFFFull);               // global_addr[31:0]
        g0[3] = (unsigned)((ga >> 32) & 0x1FFFFFFull) | (2u << 30); // [56:32] | type=2
        g1[0] = 0x00020000;                                   // data_size=4B
        g1[1] = (int)(2048u << 16);                           // tensor_dim0 = 2048
        g1[2] = (int)(1u << 16);                              // tensor_dim1 = 1
        g1[3] = (int)(2048u << 16);                           // tile_dim0 = 2048
        g1[4] = 1;                                            // tile_dim1 = 1
        g1[5] = 2048;                                         // tensor_dim0_stride
        g1[6] = 0; g1[7] = 0;
  #if defined(__clang_major__) && __clang_major__ >= 23
        v8i gz8 = {};
        __builtin_amdgcn_tensor_load_to_lds(g0, g1, gz, gz, gz8, 0);
  #else
        __builtin_amdgcn_tensor_load_to_lds(g0, g1, gz, gz, 0);
  #endif
        __builtin_amdgcn_s_wait_tensorcnt(0);
    }
    __syncthreads();
#else
    for (int i = threadIdx.x; i < 16 * HID; i += blockDim.x)
        lnF[i] = ln[(size_t)m0 * HID + i];
    __syncthreads();
#endif

    const int arl = lane & 15;
    const int akb = (lane >> 4) << 3;

    // GEMM1: [16,128]@[128,512] -> mid (+b1, relu). 32 col tiles, 4 per wave.
    for (int t = 0; t < 4; ++t) {
        const int nt = wid * 4 + t;
        v8f acc = {};
        for (int k0 = 0; k0 < HID; k0 += 32) {
            const v4f* ap4 = (const v4f*)&lnF[arl * HID + k0 + akb];
            v4f a0 = ap4[0], a1 = ap4[1], a2 = ap4[4], a3 = ap4[5];
            v16bf af;
            #pragma unroll
            for (int e = 0; e < 4; ++e) {
                af[e]      = f2bf(a0[e]);
                af[e + 4]  = f2bf(a1[e]);
                af[e + 8]  = f2bf(a2[e]);
                af[e + 12] = f2bf(a3[e]);
            }
            v16bf bfr = W1p[(size_t)((k0 >> 5) * (FFN_N / 16) + nt) * 32 + lane];
            acc = __builtin_amdgcn_wmma_f32_16x16x32_bf16(false, af, false, bfr,
                                                          (short)0, acc, false, false);
        }
        const int r0 = (lane >> 4) << 3;
        const int c  = (nt << 4) + (lane & 15);
        const float bb = b1[c];
        #pragma unroll
        for (int r = 0; r < 8; ++r)
            mid[(r0 + r) * FFN_N + c] = f2bf(fmaxf(acc[r] + bb, 0.f));
    }
    __syncthreads();

    // GEMM2: [16,512]@[512,128] -> out (+h +b2). 8 col tiles, 1 per wave.
    {
        const int nt = wid;
        v8f acc = {};
        for (int k0 = 0; k0 < FFN_N; k0 += 32) {
            const v8bf* mp = (const v8bf*)&mid[arl * FFN_N + k0 + akb];
            v8bf x0 = mp[0], x1 = mp[2];
            v16bf af;
            #pragma unroll
            for (int e = 0; e < 8; ++e) { af[e] = x0[e]; af[e + 8] = x1[e]; }
            v16bf bfr = W2p[(size_t)((k0 >> 5) * (HID / 16) + nt) * 32 + lane];
            acc = __builtin_amdgcn_wmma_f32_16x16x32_bf16(false, af, false, bfr,
                                                          (short)0, acc, false, false);
        }
        const int r0 = (lane >> 4) << 3;
        const int c  = (nt << 4) + (lane & 15);
        const float bb = b2[c];
        #pragma unroll
        for (int r = 0; r < 8; ++r) {
            size_t off = (size_t)(m0 + r0 + r) * HID + c;
            out[off] = hres[off] + acc[r] + bb;
        }
    }
}

// ---------------- small kernels ----------------

__global__ void k_fill(float* p, float v, int n) {
    int t = blockIdx.x * blockDim.x + threadIdx.x;
    if (t < n) p[t] = v;
}

__global__ void k_gather_feat(const int* __restrict__ fid, const float* __restrict__ tab,
                              float* __restrict__ feat) {
    int t = blockIdx.x * blockDim.x + threadIdx.x;
    if (t >= F_N * HID) return;
    int i = t >> 7, j = t & (HID - 1);
    feat[t] = tab[(size_t)fid[i] * HID + j];
}

__global__ void k_ui_pre(const float* __restrict__ feat, const int* __restrict__ e_src,
                         const int* __restrict__ uid, const int* __restrict__ iid,
                         const float* __restrict__ user_tab, const float* __restrict__ item_tab,
                         float* __restrict__ tmp) {
    int t = blockIdx.x * blockDim.x + threadIdx.x;
    if (t >= (U_N + I_N) * HID) return;
    int n = t >> 7, j = t & (HID - 1);
    int supRow = S_N + n;
    float s = 0.f;
    #pragma unroll
    for (int e = 0; e < DEG; ++e)
        s += feat[(size_t)e_src[supRow * DEG + e] * HID + j];
    s *= (1.f / DEG);
    float base = (n < U_N) ? user_tab[(size_t)uid[n] * HID + j]
                           : item_tab[(size_t)iid[n - U_N] * HID + j];
    tmp[t] = base + s;
}

__global__ void k_dot16(const float* __restrict__ z, const float* __restrict__ a,
                        float* __restrict__ out, int n) {
    int t = blockIdx.x * blockDim.x + threadIdx.x;
    if (t >= n * NH) return;
    int node = t >> 3, h = t & 7;
    const float* zp = z + (size_t)node * HID + h * DH;
    const float* ap = a + h * DH;
    float s = 0.f;
    #pragma unroll
    for (int e = 0; e < DH; ++e) s += zp[e] * ap[e];
    out[t] = s;
}

__global__ void k_edge_score(const int* __restrict__ src, const int* __restrict__ dst,
                             const float* __restrict__ ew, const float* __restrict__ el,
                             const float* __restrict__ er, const float* __restrict__ ae,
                             float* __restrict__ esc, float* __restrict__ m) {
    int t = blockIdx.x * blockDim.x + threadIdx.x;
    if (t >= E_N * NH) return;
    int k = t >> 3, h = t & 7;
    int s = src[k], d = dst[k];
    float e = el[s * NH + h] + er[d * NH + h] + ew[k] * ae[h];
    e = (e > 0.f) ? e : 0.2f * e;                 // leaky_relu(0.2)
    esc[t] = e;
    atomicMaxF(&m[d * NH + h], e);
}

__global__ void k_fix_m(float* __restrict__ m, int n) {
    int t = blockIdx.x * blockDim.x + threadIdx.x;
    if (t >= n) return;
    float x = m[t];
    if (!(x > -__builtin_inff())) m[t] = 0.f;
}

__global__ void k_edge_exp(const int* __restrict__ dst, float* __restrict__ esc,
                           const float* __restrict__ m, float* __restrict__ den) {
    int t = blockIdx.x * blockDim.x + threadIdx.x;
    if (t >= E_N * NH) return;
    int k = t >> 3, h = t & 7;
    int d = dst[k];
    float ex = expf(esc[t] - m[d * NH + h]);
    esc[t] = ex;
    atomicAdd(&den[d * NH + h], ex);
}

__global__ void k_edge_agg(const int* __restrict__ src, const int* __restrict__ dst,
                           const float* __restrict__ esc, const float* __restrict__ den,
                           const float* __restrict__ zsrc, float* __restrict__ agg) {
    int t = blockIdx.x * blockDim.x + threadIdx.x;
    if (t >= E_N * HID) return;
    int k = t >> 7, j = t & (HID - 1), h = j >> 4;
    int s = src[k], d = dst[k];
    float alpha = esc[k * NH + h] / fmaxf(den[d * NH + h], 1e-9f);
    atomicAdd(&agg[(size_t)d * HID + j], alpha * zsrc[(size_t)s * HID + j]);
}

__global__ void k_elu_ln(float* __restrict__ agg, const float* __restrict__ g,
                         const float* __restrict__ b, float* __restrict__ lnbuf) {
    __shared__ float red[HID];
    const int r = blockIdx.x, j = threadIdx.x;
    float x = agg[(size_t)r * HID + j];
    float h = (x > 0.f) ? x : (expf(x) - 1.f);
    red[j] = h; __syncthreads();
    for (int s = 64; s > 0; s >>= 1) { if (j < s) red[j] += red[j + s]; __syncthreads(); }
    float mu = red[0] * (1.f / HID);
    __syncthreads();
    float dv = h - mu;
    red[j] = dv * dv; __syncthreads();
    for (int s = 64; s > 0; s >>= 1) { if (j < s) red[j] += red[j + s]; __syncthreads(); }
    float var = red[0] * (1.f / HID);
    float ln = dv * rsqrtf(var + 1e-6f) * g[j] + b[j];
    agg[(size_t)r * HID + j] = h;
    lnbuf[(size_t)r * HID + j] = ln;
}

__global__ void k_head(const float* __restrict__ sup, const float* __restrict__ whW,
                       const float* __restrict__ whb, float* __restrict__ out) {
    int t = blockIdx.x * blockDim.x + threadIdx.x;
    if (t >= S_N * 2) return;
    int s = t >> 1, c = t & 1;
    const float* sp = sup + (size_t)s * HID;
    float acc = whb[c];
    #pragma unroll 4
    for (int j = 0; j < HID; ++j) acc += sp[j] * whW[j * 2 + c];
    out[t] = acc;
}

// ---------------- host orchestration ----------------

static void gemm(hipStream_t st, const float* A, const int* idxA, int K,
                 const __bf16* Bp, const float* bias, float* C, int M, int N, int act) {
    int tiles = (M / 16) * (N / 16);
    k_gemm_bf16<<<CDIV(tiles, 4), 128, 0, st>>>(A, idxA, K, (const v16bf*)Bp, N,
                                                bias, C, M, K, act);
}

static void pack_b(hipStream_t st, const float* B, __bf16* P, int K, int N) {
    int total = (K / 32) * (N / 16) * 32;
    k_pack_b<<<CDIV(total, 256), 256, 0, st>>>(B, P, K, N);
}

static void run_gat(hipStream_t st, const float* hsrc, const float* hdst,
                    int Nsrc, int Ndst, const int* srcIdx, const int* dstIdx,
                    const float* ew, const __bf16* WcatP, const float* al, const float* ar,
                    const float* ae, const __bf16* W1p, const float* b1, const __bf16* W2p,
                    const float* b2, const float* g, const float* bp, float* outBuf,
                    float* zsrc, float* zdstAgg, float* elB, float* erB,
                    float* mB, float* denB, float* escB, float* lnbuf) {
    gemm(st, hsrc, nullptr, HID, WcatP, nullptr, zsrc,    Nsrc, HID, 0);
    gemm(st, hdst, nullptr, HID, WcatP, nullptr, zdstAgg, Ndst, HID, 0);
    k_dot16<<<CDIV(Nsrc * NH, 256), 256, 0, st>>>(zsrc,    al, elB, Nsrc);
    k_dot16<<<CDIV(Ndst * NH, 256), 256, 0, st>>>(zdstAgg, ar, erB, Ndst);
    k_fill<<<CDIV(Ndst * NH, 256), 256, 0, st>>>(mB,   -__builtin_inff(), Ndst * NH);
    k_fill<<<CDIV(Ndst * NH, 256), 256, 0, st>>>(denB, 0.f, Ndst * NH);
    k_fill<<<CDIV(Ndst * HID, 256), 256, 0, st>>>(zdstAgg, 0.f, Ndst * HID); // z_dst dead, reuse as agg
    k_edge_score<<<CDIV(E_N * NH, 256), 256, 0, st>>>(srcIdx, dstIdx, ew, elB, erB, ae, escB, mB);
    k_fix_m<<<CDIV(Ndst * NH, 256), 256, 0, st>>>(mB, Ndst * NH);
    k_edge_exp<<<CDIV(E_N * NH, 256), 256, 0, st>>>(dstIdx, escB, mB, denB);
    k_edge_agg<<<CDIV(E_N * HID, 256), 256, 0, st>>>(srcIdx, dstIdx, escB, denB, zsrc, zdstAgg);
    k_elu_ln<<<Ndst, HID, 0, st>>>(zdstAgg, g, bp, lnbuf);
    k_ffn<<<Ndst / 16, 256, 0, st>>>(lnbuf, zdstAgg, (const v16bf*)W1p, b1,
                                     (const v16bf*)W2p, b2, outBuf);
}

extern "C" void kernel_launch(void* const* d_in, const int* in_sizes, int n_in,
                              void* d_out, int out_size, void* d_ws, size_t ws_size,
                              hipStream_t stream) {
    const int*   fid      = (const int*)  d_in[0];
    const int*   sid      = (const int*)  d_in[1];
    const int*   uid      = (const int*)  d_in[2];
    const int*   iid      = (const int*)  d_in[3];
    const int*   e_src    = (const int*)  d_in[4];
    const int*   e_dst    = (const int*)  d_in[5];
    const float* e_w      = (const float*)d_in[6];
    const float* feat_tab = (const float*)d_in[7];
    const float* sent_tab = (const float*)d_in[8];
    const float* user_tab = (const float*)d_in[9];
    const float* item_tab = (const float*)d_in[10];
    const float* Wsp      = (const float*)d_in[11];
    const float* bsp      = (const float*)d_in[12];
    const float* Wup      = (const float*)d_in[13];
    const float* Wip      = (const float*)d_in[14];
    const float* w2s[10]; for (int i = 0; i < 10; ++i) w2s[i] = (const float*)d_in[15 + i];
    const float* s2w[10]; for (int i = 0; i < 10; ++i) s2w[i] = (const float*)d_in[25 + i];
    const float* whW = (const float*)d_in[35];
    const float* whb = (const float*)d_in[36];
    float* out = (float*)d_out;

    float* ws = (float*)d_ws;
    size_t o = 0;
    float* featB   = ws + o; o += (size_t)F_N * HID;
    float* featN   = ws + o; o += (size_t)F_N * HID;
    float* supB    = ws + o; o += (size_t)NSUP_N * HID;
    float* supN    = ws + o; o += (size_t)NSUP_N * HID;
    float* zsrcB   = ws + o; o += (size_t)NSUP_N * HID;
    float* zdstB   = ws + o; o += (size_t)NSUP_N * HID;
    float* lnB     = ws + o; o += (size_t)NSUP_N * HID;
    float* elB     = ws + o; o += (size_t)NSUP_N * NH;
    float* erB     = ws + o; o += (size_t)NSUP_N * NH;
    float* mB      = ws + o; o += (size_t)NSUP_N * NH;
    float* denB    = ws + o; o += (size_t)NSUP_N * NH;
    float* escB    = ws + o; o += (size_t)E_N * NH;
    float* tmpUI   = ws + o; o += (size_t)(U_N + I_N) * HID;
    // packed bf16 weights (each n bf16 elems consumes n/2 floats; all sizes even)
    __bf16* WspP   = (__bf16*)(ws + o); o += (size_t)SE_N * HID / 2;
    __bf16* WupP   = (__bf16*)(ws + o); o += (size_t)HID * HID / 2;
    __bf16* WipP   = (__bf16*)(ws + o); o += (size_t)HID * HID / 2;
    __bf16* WcAP   = (__bf16*)(ws + o); o += (size_t)HID * HID / 2;   // w2s Wcat
    __bf16* WcBP   = (__bf16*)(ws + o); o += (size_t)HID * HID / 2;   // s2w Wcat
    __bf16* W1aP   = (__bf16*)(ws + o); o += (size_t)HID * FFN_N / 2; // w2s W1
    __bf16* W2aP   = (__bf16*)(ws + o); o += (size_t)FFN_N * HID / 2; // w2s W2
    __bf16* W1bP   = (__bf16*)(ws + o); o += (size_t)HID * FFN_N / 2; // s2w W1
    __bf16* W2bP   = (__bf16*)(ws + o); o += (size_t)FFN_N * HID / 2; // s2w W2
    (void)ws_size; (void)in_sizes; (void)n_in; (void)out_size;

    // ---- pack weights to bf16 fragment order (once per launch, tiny) ----
    pack_b(stream, Wsp,    WspP, SE_N, HID);
    pack_b(stream, Wup,    WupP, HID,  HID);
    pack_b(stream, Wip,    WipP, HID,  HID);
    pack_b(stream, w2s[4], W1aP, HID,  FFN_N);
    pack_b(stream, w2s[6], W2aP, FFN_N, HID);
    pack_b(stream, s2w[4], W1bP, HID,  FFN_N);
    pack_b(stream, s2w[6], W2bP, FFN_N, HID);
    k_pack_Whead<<<CDIV(1024, 256), 256, 0, stream>>>(w2s[0], WcAP);
    k_pack_Whead<<<CDIV(1024, 256), 256, 0, stream>>>(s2w[0], WcBP);

    // ---- init embeddings ----
    k_gather_feat<<<CDIV(F_N * HID, 256), 256, 0, stream>>>(fid, feat_tab, featB);
    gemm(stream, sent_tab, sid, SE_N, WspP, bsp, supB, S_N, HID, 0);
    k_ui_pre<<<CDIV((U_N + I_N) * HID, 256), 256, 0, stream>>>(featB, e_src, uid, iid,
                                                               user_tab, item_tab, tmpUI);
    gemm(stream, tmpUI,                     nullptr, HID, WupP, nullptr,
         supB + (size_t)S_N * HID,         U_N, HID, 0);
    gemm(stream, tmpUI + (size_t)U_N * HID, nullptr, HID, WipP, nullptr,
         supB + (size_t)(S_N + U_N) * HID, I_N, HID, 0);

    // ---- GAT stack (NITER = 1) ----
    run_gat(stream, featB, supB, F_N, NSUP_N, e_src, e_dst, e_w,
            WcAP, w2s[1], w2s[2], w2s[3], W1aP, w2s[5], W2aP, w2s[7], w2s[8], w2s[9],
            supN, zsrcB, zdstB, elB, erB, mB, denB, escB, lnB);
    run_gat(stream, supN, featB, NSUP_N, F_N, e_dst, e_src, e_w,
            WcBP, s2w[1], s2w[2], s2w[3], W1bP, s2w[5], W2bP, s2w[7], s2w[8], s2w[9],
            featN, zsrcB, zdstB, elB, erB, mB, denB, escB, lnB);
    run_gat(stream, featN, supN, F_N, NSUP_N, e_src, e_dst, e_w,
            WcAP, w2s[1], w2s[2], w2s[3], W1aP, w2s[5], W2aP, w2s[7], w2s[8], w2s[9],
            supB, zsrcB, zdstB, elB, erB, mB, denB, escB, lnB);

    // ---- head: sup2[:S] @ whW + whb ----
    k_head<<<CDIV(S_N * 2, 256), 256, 0, stream>>>(supB, whW, whb, out);
}
